// TransformerBlock_44143673868717
// MI455X (gfx1250) — matrix operand order repaired and checked by reading, not compiled, not run on hardware
//
#include <hip/hip_runtime.h>
#include <hip/hip_bf16.h>
#include <cmath>

// Problem constants
#define Bb   2
#define Ss   2048
#define Dd   768
#define PDd  128
#define Hh   8
#define RFf  16
#define DQq  128
#define DKk  128
#define DVv  192
#define Pp   128          // S / RF
#define MTOK (Bb*Ss)      // 4096 tokens
#define HD   (Hh*DQq)     // 1024
#define HDV  (Hh*DVv)     // 1536
#define DFF  (2*Dd)       // 1536
#define RMSC 0.9999995000003750f   // 1/sqrt(1+1e-6)

typedef __attribute__((ext_vector_type(16))) _Float16 v16h;
typedef __attribute__((ext_vector_type(8)))  float    v8f;

union FragH { v16h h; uint4 u[2]; };

__device__ __forceinline__ v8f wmma_f16(const FragH& a, const FragH& b, v8f c) {
  // D = A(16x32 f16) * B(32x16 f16) + C(16x16 f32)
  return __builtin_amdgcn_wmma_f32_16x16x32_f16(false, a.h, false, b.h, (short)0, c,
                                                false, false);
}

// ---- CDNA5 async global->LDS copy (ASYNCcnt) with sync fallback ----
#if __has_builtin(__builtin_amdgcn_global_load_async_to_lds_b128)
#define HAVE_ASYNC_LDS 1
#else
#define HAVE_ASYNC_LDS 0
#endif

typedef int v4i_vec __attribute__((vector_size(16)));
typedef __attribute__((address_space(1))) v4i_vec* gptr4;   // global int4*
typedef __attribute__((address_space(3))) v4i_vec* lptr4;   // LDS int4*

__device__ __forceinline__ void cp16_issue(_Float16* lds_dst, const _Float16* gsrc) {
#if HAVE_ASYNC_LDS
  void* gv = const_cast<_Float16*>(gsrc);
  void* lv = lds_dst;
  __builtin_amdgcn_global_load_async_to_lds_b128((gptr4)gv, (lptr4)lv, 0, 0);
#else
  *(uint4*)lds_dst = *(const uint4*)gsrc;
#endif
}
__device__ __forceinline__ void cp_commit_wait() {
#if HAVE_ASYNC_LDS
  asm volatile("s_wait_asynccnt 0" ::: "memory");
#endif
}

__device__ __forceinline__ float wred_max16(float v) {
  #pragma unroll
  for (int m = 1; m < 16; m <<= 1) v = fmaxf(v, __shfl_xor(v, m, 32));
  return v;
}
__device__ __forceinline__ float wred_sum16(float v) {
  #pragma unroll
  for (int m = 1; m < 16; m <<= 1) v += __shfl_xor(v, m, 32);
  return v;
}
__device__ __forceinline__ float wred_sum32(float v) {
  #pragma unroll
  for (int m = 1; m < 32; m <<= 1) v += __shfl_xor(v, m, 32);
  return v;
}

// ---------------------------------------------------------------- utilities
__global__ void copy_f32(const float* __restrict__ a, float* __restrict__ o, int n) {
  int i = blockIdx.x * blockDim.x + threadIdx.x;
  if (i < n) o[i] = a[i];
}

// out[i] = (f16)(g[d] * x[i] * RMSC + b[d])   (rms_eval, eval mode == affine)
__global__ void affine_h_kernel(const float* __restrict__ x, const float* __restrict__ g,
                                const float* __restrict__ b, _Float16* __restrict__ out,
                                int n, int Dc) {
  int i = blockIdx.x * blockDim.x + threadIdx.x;
  if (i < n) {
    int d = i % Dc;
    out[i] = (_Float16)(g[d] * x[i] * RMSC + b[d]);
  }
}

// W (K x N f32) -> WT (N x K f16)
__global__ void wconvT(const float* __restrict__ W, _Float16* __restrict__ WT,
                       int K, int N) {
  int i = blockIdx.x * blockDim.x + threadIdx.x;
  if (i < K * N) {
    int n = i / K, k = i % K;
    WT[i] = (_Float16)W[k * N + n];
  }
}

// ------------------------------------------------ pair bias (B,H,P,P) compact
// wave per (b,i,j): h = gelu(rms_eval(pair_x)); bias[b,hh,i,j] = h . W[:,hh]
__global__ __launch_bounds__(256) void bias_kernel(const float* __restrict__ px,
                                                   const float* __restrict__ gg,
                                                   const float* __restrict__ bb,
                                                   const float* __restrict__ W,
                                                   float* __restrict__ biasc) {
  int lane = threadIdx.x & 31;
  int row  = blockIdx.x * 8 + (threadIdx.x >> 5);   // over B*P*P
  int j = row % Pp;
  int i = (row / Pp) % Pp;
  int b = row / (Pp * Pp);
  const float* src = px + (size_t)row * PDd;
  float hv[4];
  #pragma unroll
  for (int t = 0; t < 4; t++) {
    int d = lane + 32 * t;
    float x = gg[d] * src[d] * RMSC + bb[d];
    hv[t] = 0.5f * x * (1.f + erff(x * 0.70710678118654752f)); // exact gelu
  }
  #pragma unroll
  for (int hh = 0; hh < Hh; hh++) {
    float s = 0.f;
    #pragma unroll
    for (int t = 0; t < 4; t++) s += hv[t] * W[(lane + 32 * t) * Hh + hh];
    s = wred_sum32(s);
    if (lane == 0) biasc[(((size_t)b * Hh + hh) * Pp + i) * Pp + j] = s;
  }
}

// ---------------------------------------------------------------- WMMA GEMM
// A: (M x K) f16 row-major.  BT: (N x K) f16 row-major (i.e. B transposed).
// Block tile 128x64; 8 waves of 32 threads, each wave a 32x32 subtile.
// Double-buffered LDS staging via async global->LDS (ASYNCcnt).
// EP 0: outF = A@B       (raw f32)
// EP 1: outH = f16(relu(A@B + bias))
// EP 2: resid += eg*((A@B + bias)*RMSC) + eb     (residual + rms_eval)
template <int EP>
__global__ __launch_bounds__(256) void gemm_f16(
    const _Float16* __restrict__ A, const _Float16* __restrict__ BT,
    int M, int N, int K,
    float* __restrict__ outF, _Float16* __restrict__ outH, float* __restrict__ resid,
    const float* __restrict__ bias, const float* __restrict__ eg,
    const float* __restrict__ eb) {
  __shared__ __align__(16) _Float16 As[2][128 * 32];
  __shared__ __align__(16) _Float16 Bs[2][64 * 32];
  int t = threadIdx.x;
  int lane = t & 31, w = t >> 5;
  int wm = w & 3, wn = w >> 2;
  int lm = lane & 15, g = lane >> 4;
  int m0 = blockIdx.x * 128, n0 = blockIdx.y * 64;

  v8f acc[2][2];
  #pragma unroll
  for (int a = 0; a < 2; a++)
    #pragma unroll
    for (int c = 0; c < 2; c++)
      acc[a][c] = (v8f){0.f, 0.f, 0.f, 0.f, 0.f, 0.f, 0.f, 0.f};

  int frow = t >> 2, fcol = (t & 3) * 8;     // 256 threads: 64 rows x 4 chunks
  const _Float16* gA0 = &A[(size_t)(m0 + frow) * K + fcol];
  const _Float16* gA1 = &A[(size_t)(m0 + frow + 64) * K + fcol];
  const _Float16* gB  = &BT[(size_t)(n0 + frow) * K + fcol];

  auto fill = [&](int buf, int k0) {
    cp16_issue(&As[buf][frow * 32 + fcol],        gA0 + k0);
    cp16_issue(&As[buf][(frow + 64) * 32 + fcol], gA1 + k0);
    cp16_issue(&Bs[buf][frow * 32 + fcol],        gB + k0);
  };

  fill(0, 0);
  cp_commit_wait();
  __syncthreads();

  const int nk = K / 32;
  for (int ki = 0; ki < nk; ki++) {
    int cur = ki & 1, nxt = cur ^ 1;
    if (ki + 1 < nk) fill(nxt, (ki + 1) * 32);   // prefetch next tile (async)

    FragH af[2], bf[2];
    #pragma unroll
    for (int i = 0; i < 2; i++) {
      int ar = wm * 32 + i * 16 + lm;
      af[i].u[0] = *(const uint4*)&As[cur][ar * 32 + g * 8];
      af[i].u[1] = *(const uint4*)&As[cur][ar * 32 + 16 + g * 8];
      int bc = wn * 32 + i * 16 + lm;
      bf[i].u[0] = *(const uint4*)&Bs[cur][bc * 32 + g * 8];
      bf[i].u[1] = *(const uint4*)&Bs[cur][bc * 32 + 16 + g * 8];
    }
    #pragma unroll
    for (int mt = 0; mt < 2; mt++)
      #pragma unroll
      for (int nt = 0; nt < 2; nt++)
        acc[mt][nt] = wmma_f16(af[mt], bf[nt], acc[mt][nt]);

    cp_commit_wait();     // next tile resident
    __syncthreads();      // everyone done with cur, fills visible
  }

  #pragma unroll
  for (int mt = 0; mt < 2; mt++)
    #pragma unroll
    for (int nt = 0; nt < 2; nt++)
      #pragma unroll
      for (int r = 0; r < 8; r++) {
        int grow = m0 + wm * 32 + mt * 16 + r + 8 * g;
        int gcol = n0 + wn * 32 + nt * 16 + lm;
        float v = acc[mt][nt][r];
        if constexpr (EP == 0) {
          outF[(size_t)grow * N + gcol] = v;
        } else if constexpr (EP == 1) {
          outH[(size_t)grow * N + gcol] = (_Float16)fmaxf(v + bias[gcol], 0.f);
        } else {
          resid[(size_t)grow * N + gcol] +=
              eg[gcol] * ((v + bias[gcol]) * RMSC) + eb[gcol];
        }
      }
}

// ---------------------------------------- per-row LayerNorm (+RoPE) kernels
// Q: rows over (b,h,s); src q_f32 (B*S, H*DQ); dst q_h (B,H,S,DQ) f16, scaled 1/sqrt(DK)
__global__ __launch_bounds__(256) void q_ln_rope(const float* __restrict__ qf,
                                                 const float* __restrict__ gq,
                                                 const float* __restrict__ bq,
                                                 _Float16* __restrict__ qh) {
  int lane = threadIdx.x & 31;
  int row  = blockIdx.x * 8 + (threadIdx.x >> 5);  // over B*H*S
  int s = row % Ss;
  int bh = row / Ss;
  int b = bh / Hh, h = bh % Hh;
  const float* src = qf + ((size_t)(b * Ss + s)) * HD + h * DQq;
  float v[4], sum = 0.f;
  #pragma unroll
  for (int t = 0; t < 4; t++) { v[t] = src[lane + 32 * t]; sum += v[t]; }
  float mean = wred_sum32(sum) * (1.f / 128.f);
  float var = 0.f;
  #pragma unroll
  for (int t = 0; t < 4; t++) { float d = v[t] - mean; var += d * d; }
  var = wred_sum32(var) * (1.f / 128.f);
  float rstd = rsqrtf(var + 1e-5f);
  #pragma unroll
  for (int t = 0; t < 4; t++) {
    int d = lane + 32 * t;
    v[t] = gq[d] * (v[t] - mean) * rstd + bq[d];
  }
  float a0 = (float)s * powf(8192.f, -(float)lane / 64.f);
  float a1 = (float)s * powf(8192.f, -(float)(lane + 32) / 64.f);
  float c0 = cosf(a0), s0 = sinf(a0), c1 = cosf(a1), s1 = sinf(a1);
  const float sc = 0.08838834764831845f;  // 1/sqrt(128) baked into Q
  _Float16* dst = qh + ((size_t)bh * Ss + s) * DQq;
  dst[lane]      = (_Float16)((v[0] * c0 - v[2] * s0) * sc);
  dst[lane + 32] = (_Float16)((v[1] * c1 - v[3] * s1) * sc);
  dst[lane + 64] = (_Float16)((v[0] * s0 + v[2] * c0) * sc);
  dst[lane + 96] = (_Float16)((v[1] * s1 + v[3] * c1) * sc);
}

// K: rows over (b,s); src (B*S, DK); dst k_h (B,S,DK)
__global__ __launch_bounds__(256) void k_ln_rope(const float* __restrict__ kf,
                                                 const float* __restrict__ gk,
                                                 const float* __restrict__ bk,
                                                 _Float16* __restrict__ kh) {
  int lane = threadIdx.x & 31;
  int row  = blockIdx.x * 8 + (threadIdx.x >> 5);  // over B*S
  int s = row % Ss;
  const float* src = kf + (size_t)row * DKk;
  float v[4], sum = 0.f;
  #pragma unroll
  for (int t = 0; t < 4; t++) { v[t] = src[lane + 32 * t]; sum += v[t]; }
  float mean = wred_sum32(sum) * (1.f / 128.f);
  float var = 0.f;
  #pragma unroll
  for (int t = 0; t < 4; t++) { float d = v[t] - mean; var += d * d; }
  var = wred_sum32(var) * (1.f / 128.f);
  float rstd = rsqrtf(var + 1e-5f);
  #pragma unroll
  for (int t = 0; t < 4; t++) {
    int d = lane + 32 * t;
    v[t] = gk[d] * (v[t] - mean) * rstd + bk[d];
  }
  float a0 = (float)s * powf(8192.f, -(float)lane / 64.f);
  float a1 = (float)s * powf(8192.f, -(float)(lane + 32) / 64.f);
  float c0 = cosf(a0), s0 = sinf(a0), c1 = cosf(a1), s1 = sinf(a1);
  _Float16* dst = kh + (size_t)row * DKk;
  dst[lane]      = (_Float16)(v[0] * c0 - v[2] * s0);
  dst[lane + 32] = (_Float16)(v[1] * c1 - v[3] * s1);
  dst[lane + 64] = (_Float16)(v[0] * s0 + v[2] * c0);
  dst[lane + 96] = (_Float16)(v[1] * s1 + v[3] * c1);
}

// V: rows over (b,s); src (B*S, DV); dst v_t TRANSPOSED (B, DV, S) f16
__global__ __launch_bounds__(256) void v_ln(const float* __restrict__ vf,
                                            const float* __restrict__ gv,
                                            const float* __restrict__ bv,
                                            _Float16* __restrict__ vt) {
  int lane = threadIdx.x & 31;
  int row  = blockIdx.x * 8 + (threadIdx.x >> 5);  // over B*S
  int s = row % Ss;
  int b = row / Ss;
  const float* src = vf + (size_t)row * DVv;
  float v[6], sum = 0.f;
  #pragma unroll
  for (int t = 0; t < 6; t++) { v[t] = src[lane + 32 * t]; sum += v[t]; }
  float mean = wred_sum32(sum) * (1.f / 192.f);
  float var = 0.f;
  #pragma unroll
  for (int t = 0; t < 6; t++) { float d = v[t] - mean; var += d * d; }
  var = wred_sum32(var) * (1.f / 192.f);
  float rstd = rsqrtf(var + 1e-5f);
  #pragma unroll
  for (int t = 0; t < 6; t++) {
    int d = lane + 32 * t;
    vt[((size_t)b * DVv + d) * Ss + s] = (_Float16)(gv[d] * (v[t] - mean) * rstd + bv[d]);
  }
}

// ------------------------------------------------------- flash attention
// grid: B*H*(S/64) blocks, 128 threads (4 waves); each wave = 16 query rows.
// K/V tiles cooperatively double-buffered in LDS (async global->LDS),
// shared by all 4 waves. logits = q.k + bias_tile; cap 5*tanh(l/5);
// online softmax; O = P@V.
__global__ __launch_bounds__(128) void attn_kernel(
    const _Float16* __restrict__ qh,   // (B,H,S,DQ), pre-scaled by 1/sqrt(DK)
    const _Float16* __restrict__ kh,   // (B,S,DK)
    const _Float16* __restrict__ vt,   // (B,DV,S)
    const float* __restrict__ biasc,   // (B,H,P,P)
    _Float16* __restrict__ yh) {       // (B,S,H*DV)
  __shared__ __align__(16) _Float16 Ks[2][32 * 128];   // 32 keys x DK
  __shared__ __align__(16) _Float16 Vs[2][192 * 32];   // DV x 32 keys (col-major keys)
  __shared__ __align__(16) _Float16 ldsp[4][16 * 32];  // per-wave P staging
  int t = threadIdx.x;
  int lane = t & 31, w = t >> 5;
  int lm = lane & 15, g = lane >> 4;
  const int nsb = Ss / 64;
  int bh = blockIdx.x / nsb;
  int sb = blockIdx.x % nsb;
  int b = bh / Hh, h = bh % Hh;
  int q0 = sb * 64 + w * 16;

  const _Float16* kbase = kh + (size_t)b * Ss * DKk;
  const _Float16* vbase = vt + (size_t)b * DVv * Ss;

  auto fill = [&](int buf, int j0) {
    #pragma unroll
    for (int r = 0; r < 4; r++) {            // K tile: 512 x 16B
      int idx = t + 128 * r;
      int row = idx >> 4, col = (idx & 15) * 8;
      cp16_issue(&Ks[buf][row * 128 + col],
                 kbase + (size_t)(j0 + row) * DKk + col);
    }
    #pragma unroll
    for (int r = 0; r < 6; r++) {            // V tile: 768 x 16B
      int idx = t + 128 * r;
      int row = idx >> 2, col = (idx & 3) * 8;
      cp16_issue(&Vs[buf][row * 32 + col],
                 vbase + (size_t)row * Ss + j0 + col);
    }
  };

  // Q tile: 4 A-fragments (16x32 each) held in registers for the whole loop
  FragH qf[4];
  const _Float16* qbase = qh + ((size_t)bh * Ss + q0 + lm) * DQq;
  #pragma unroll
  for (int f = 0; f < 4; f++) {
    qf[f].u[0] = *(const uint4*)&qbase[f * 32 + g * 8];
    qf[f].u[1] = *(const uint4*)&qbase[f * 32 + 16 + g * 8];
  }

  float mrow[8], lrow[8];
  v8f o[12];
  #pragma unroll
  for (int r = 0; r < 8; r++) { mrow[r] = -1e30f; lrow[r] = 0.f; }
  #pragma unroll
  for (int f = 0; f < 12; f++) o[f] = (v8f){0.f, 0.f, 0.f, 0.f, 0.f, 0.f, 0.f, 0.f};

  const float* brow = biasc + ((size_t)bh * Pp + (q0 >> 4)) * Pp;

  fill(0, 0);
  cp_commit_wait();
  __syncthreads();

  for (int j0 = 0; j0 < Ss; j0 += 32) {
    int cur = (j0 >> 5) & 1, nxt = cur ^ 1;
    if (j0 + 32 < Ss) fill(nxt, j0 + 32);    // async prefetch of next K/V tile

    // ---- logits: two 16x16 tiles (columns j0..j0+31) from Ks[cur]
    v8f sc[2];
    #pragma unroll
    for (int n = 0; n < 2; n++) {
      sc[n] = (v8f){0.f, 0.f, 0.f, 0.f, 0.f, 0.f, 0.f, 0.f};
      const _Float16* kb = &Ks[cur][(n * 16 + lm) * 128];
      #pragma unroll
      for (int f = 0; f < 4; f++) {
        FragH kfr;
        kfr.u[0] = *(const uint4*)&kb[f * 32 + g * 8];
        kfr.u[1] = *(const uint4*)&kb[f * 32 + 16 + g * 8];
        sc[n] = wmma_f16(qf[f], kfr, sc[n]);
      }
      float bvl = brow[(j0 >> 4) + n];      // bias constant over each 16x16 tile
      #pragma unroll
      for (int r = 0; r < 8; r++)
        sc[n][r] = 5.f * tanhf(0.2f * (sc[n][r] + bvl));
    }
    // ---- online softmax (rows r+8g, cols striped on 16 lanes)
    #pragma unroll
    for (int r = 0; r < 8; r++) {
      float mx = wred_max16(fmaxf(sc[0][r], sc[1][r]));
      float mnew = fmaxf(mrow[r], mx);
      float rsc = __expf(mrow[r] - mnew);
      float p0 = __expf(sc[0][r] - mnew);
      float p1 = __expf(sc[1][r] - mnew);
      float rsum = wred_sum16(p0 + p1);
      lrow[r] = lrow[r] * rsc + rsum;
      mrow[r] = mnew;
      sc[0][r] = p0; sc[1][r] = p1;
      #pragma unroll
      for (int f = 0; f < 12; f++) o[f][r] *= rsc;
    }
    // ---- stage P (C layout -> A layout) through per-wave LDS
    _Float16* lp = ldsp[w];
    #pragma unroll
    for (int r = 0; r < 8; r++) {
      lp[(r + 8 * g) * 32 + lm]      = (_Float16)sc[0][r];
      lp[(r + 8 * g) * 32 + 16 + lm] = (_Float16)sc[1][r];
    }
    asm volatile("s_wait_dscnt 0" ::: "memory");
    FragH pf;
    pf.u[0] = *(const uint4*)&lp[lm * 32 + g * 8];
    pf.u[1] = *(const uint4*)&lp[lm * 32 + 16 + g * 8];
    // ---- O += P @ V  from Vs[cur] (table-order B fragments)
    #pragma unroll
    for (int f = 0; f < 12; f++) {
      FragH vfr;
      const _Float16* vr = &Vs[cur][(f * 16 + lm) * 32];
      vfr.u[0] = *(const uint4*)&vr[g * 8];
      vfr.u[1] = *(const uint4*)&vr[16 + g * 8];
      o[f] = wmma_f16(pf, vfr, o[f]);
    }

    cp_commit_wait();     // next K/V tile resident
    __syncthreads();
  }
  // ---- normalize and write y (B,S,H*DV)
  #pragma unroll
  for (int r = 0; r < 8; r++) {
    float inv = 1.f / lrow[r];
    size_t row = (size_t)b * Ss + q0 + r + 8 * g;
    _Float16* yr = yh + row * HDV + h * DVv + lm;
    #pragma unroll
    for (int f = 0; f < 12; f++) yr[f * 16] = (_Float16)(o[f][r] * inv);
  }
}

// ---------------------------------------------------------------- launcher
extern "C" void kernel_launch(void* const* d_in, const int* in_sizes, int n_in,
                              void* d_out, int out_size, void* d_ws, size_t ws_size,
                              hipStream_t stream) {
  (void)in_sizes; (void)n_in; (void)out_size; (void)ws_size;
  const float* x_in   = (const float*)d_in[0];
  const float* pair_x = (const float*)d_in[1];
  auto PAR = [&](int l, int idx) -> const float* {
    return (const float*)d_in[2 + l * 26 + idx];
  };
  // per-layer param indices:
  // 0 bias_g 1 bias_b 2 bias_W 3 pre_g 4 pre_b 5 Wq 6 Wk 7 Wv 8 qln_g 9 qln_b
  // 10 kln_g 11 kln_b 12 vln_g 13 vln_b 14 Wo 15 bo 16 post_g 17 post_b
  // 18 m1_g 19 m1_b 20 W1 21 b1 22 W2 23 b2 24 m2_g 25 m2_b

  size_t off = 0;
  auto alloc = [&](size_t bytes) -> void* {
    void* p = (char*)d_ws + off;
    off += (bytes + 255) & ~(size_t)255;
    return p;
  };
  float*     x_buf = (float*)    alloc((size_t)MTOK * Dd * 4);
  _Float16*  xn_h  = (_Float16*) alloc((size_t)MTOK * Dd * 2);
  float*     q_f32 = (float*)    alloc((size_t)MTOK * HD * 4);
  float*     k_f32 = (float*)    alloc((size_t)MTOK * DKk * 4);
  float*     v_f32 = (float*)    alloc((size_t)MTOK * DVv * 4);
  _Float16*  q_h   = (_Float16*) alloc((size_t)Bb * Hh * Ss * DQq * 2);
  _Float16*  k_h   = (_Float16*) alloc((size_t)Bb * Ss * DKk * 2);
  _Float16*  v_t   = (_Float16*) alloc((size_t)Bb * DVv * Ss * 2);
  float*     biasc = (float*)    alloc((size_t)Bb * Hh * Pp * Pp * 4);
  _Float16*  y_h   = (_Float16*) alloc((size_t)MTOK * HDV * 2);
  _Float16*  h_h   = (_Float16*) alloc((size_t)MTOK * DFF * 2);
  _Float16*  wqT   = (_Float16*) alloc((size_t)HD  * Dd  * 2);
  _Float16*  wkT   = (_Float16*) alloc((size_t)DKk * Dd  * 2);
  _Float16*  wvT   = (_Float16*) alloc((size_t)DVv * Dd  * 2);
  _Float16*  woT   = (_Float16*) alloc((size_t)Dd  * HDV * 2);
  _Float16*  w1T   = (_Float16*) alloc((size_t)DFF * Dd  * 2);
  _Float16*  w2T   = (_Float16*) alloc((size_t)Dd  * DFF * 2);

  const int nXD = MTOK * Dd;
  copy_f32<<<(nXD + 255) / 256, 256, 0, stream>>>(x_in, x_buf, nXD);

  for (int l = 0; l < 2; l++) {
    // pair bias
    bias_kernel<<<Bb * Pp * Pp / 8, 256, 0, stream>>>(
        pair_x, PAR(l, 0), PAR(l, 1), PAR(l, 2), biasc);
    // pre-norm -> f16
    affine_h_kernel<<<(nXD + 255) / 256, 256, 0, stream>>>(
        x_buf, PAR(l, 3), PAR(l, 4), xn_h, nXD, Dd);
    // weights -> f16 transposed (N x K)
    wconvT<<<(HD  * Dd + 255) / 256, 256, 0, stream>>>(PAR(l, 5),  wqT, Dd,  HD);
    wconvT<<<(DKk * Dd + 255) / 256, 256, 0, stream>>>(PAR(l, 6),  wkT, Dd,  DKk);
    wconvT<<<(DVv * Dd + 255) / 256, 256, 0, stream>>>(PAR(l, 7),  wvT, Dd,  DVv);
    wconvT<<<(Dd * HDV + 255) / 256, 256, 0, stream>>>(PAR(l, 14), woT, HDV, Dd);
    wconvT<<<(DFF * Dd + 255) / 256, 256, 0, stream>>>(PAR(l, 20), w1T, Dd,  DFF);
    wconvT<<<(Dd * DFF + 255) / 256, 256, 0, stream>>>(PAR(l, 22), w2T, DFF, Dd);
    // QKV projections
    gemm_f16<0><<<dim3(MTOK / 128, HD / 64), 256, 0, stream>>>(
        xn_h, wqT, MTOK, HD, Dd, q_f32, nullptr, nullptr, nullptr, nullptr, nullptr);
    gemm_f16<0><<<dim3(MTOK / 128, DKk / 64), 256, 0, stream>>>(
        xn_h, wkT, MTOK, DKk, Dd, k_f32, nullptr, nullptr, nullptr, nullptr, nullptr);
    gemm_f16<0><<<dim3(MTOK / 128, DVv / 64), 256, 0, stream>>>(
        xn_h, wvT, MTOK, DVv, Dd, v_f32, nullptr, nullptr, nullptr, nullptr, nullptr);
    // layernorm + rope
    q_ln_rope<<<Bb * Hh * Ss / 8, 256, 0, stream>>>(q_f32, PAR(l, 8), PAR(l, 9), q_h);
    k_ln_rope<<<Bb * Ss / 8, 256, 0, stream>>>(k_f32, PAR(l, 10), PAR(l, 11), k_h);
    v_ln<<<Bb * Ss / 8, 256, 0, stream>>>(v_f32, PAR(l, 12), PAR(l, 13), v_t);
    // flash attention
    attn_kernel<<<Bb * Hh * (Ss / 64), 128, 0, stream>>>(q_h, k_h, v_t, biasc, y_h);
    // output projection + residual + post rms
    gemm_f16<2><<<dim3(MTOK / 128, Dd / 64), 256, 0, stream>>>(
        y_h, woT, MTOK, Dd, HDV, nullptr, nullptr, x_buf,
        PAR(l, 15), PAR(l, 16), PAR(l, 17));
    // MLP
    affine_h_kernel<<<(nXD + 255) / 256, 256, 0, stream>>>(
        x_buf, PAR(l, 18), PAR(l, 19), xn_h, nXD, Dd);
    gemm_f16<1><<<dim3(MTOK / 128, DFF / 64), 256, 0, stream>>>(
        xn_h, w1T, MTOK, DFF, Dd, nullptr, h_h, nullptr,
        PAR(l, 21), nullptr, nullptr);
    gemm_f16<2><<<dim3(MTOK / 128, Dd / 64), 256, 0, stream>>>(
        h_h, w2T, MTOK, Dd, DFF, nullptr, nullptr, x_buf,
        PAR(l, 23), PAR(l, 24), PAR(l, 25));
  }
  copy_f32<<<(nXD + 255) / 256, 256, 0, stream>>>(x_buf, (float*)d_out, nXD);
}
